// MessagePassingEncoder_87222195847593
// MI455X (gfx1250) — matrix operand order repaired and checked by reading, not compiled, not run on hardware
//
#include <hip/hip_runtime.h>

// ---------------------------------------------------------------------------
// MeshGraphNets-style GNN forward for MI455X (gfx1250), wave32 + WMMA f16.
// All GEMMs are [rows,K] @ [K,128]; fused 3-layer MLP per kernel with LDS
// staging, WMMA 16x16x32 f16 (fp32 accum), fused LayerNorm/residual/scatter.
// Round 2: fast-approx SiLU (v_rcp) + parallel LN reduction to cut VALU tail.
// ---------------------------------------------------------------------------

#define HID 128
#define BROWS 32   // rows per block (2 M-tiles of 16)

typedef __attribute__((ext_vector_type(16))) _Float16 v16h;
typedef __attribute__((ext_vector_type(8)))  float    v8f;

__device__ __forceinline__ v8f wmma16(v16h a, v16h b, v8f c) {
  // D = A(16x32 f16) * B(32x16 f16) + C(16x16 f32)
  return __builtin_amdgcn_wmma_f32_16x16x32_f16(false, a, false, b, (short)0, c,
                                                false, false);
}

// silu(x) = x * sigmoid(x), approximate rcp (result is rounded to f16 anyway)
__device__ __forceinline__ float fast_silu(float x) {
  return x * __builtin_amdgcn_rcpf(1.f + __expf(-x));
}

// A-fragment (16x32 f16) from LDS row-major [row][k], stride in halves.
// Contiguous k order used here (matches B swizzle): lanes 0-15 -> row=lane, K {0..15}; lanes 16-31 -> row=lane-16,
// K {16..31}.
__device__ __forceinline__ v16h load_a_frag(const _Float16* sA, int stride,
                                            int row0, int kt, int lane) {
  const _Float16* p = sA + (size_t)(row0 + (lane & 15)) * stride + kt * 32 +
                      ((lane & 16) ? 16 : 0);
  v16h a;
#pragma unroll
  for (int j = 0; j < 8; ++j) {
    a[j]     = p[j];
    a[j + 8] = p[j + 8];
  }
  return a;
}

// One 16-wide N-tile x 32-row GEMM over KT K-tiles. Weights pre-swizzled so the
// B-fragment for (nt,kt) is 32 lanes x 16 contiguous halves.
template <int KT>
__device__ __forceinline__ void gemm2(const _Float16* sA, int stride,
                                      const _Float16* W, int lane, int nt,
                                      v8f& c0, v8f& c1) {
#pragma unroll
  for (int kt = 0; kt < KT; ++kt) {
    v16h b = *(const v16h*)(W + ((size_t)(nt * KT + kt) * 32 + lane) * 16);
    v16h a0 = load_a_frag(sA, stride, 0, kt, lane);
    c0 = wmma16(a0, b, c0);
    v16h a1 = load_a_frag(sA, stride, 16, kt, lane);
    c1 = wmma16(a1, b, c1);
  }
}

// C/D layout: lane 0-15: N=lane, M=v; lane 16-31: N=lane-16, M=8+v.
__device__ __forceinline__ void store_silu(_Float16* sOut, int lane, int nt,
                                           v8f c0, v8f c1, float bias) {
  int n = nt * 16 + (lane & 15);
  int mb = (lane & 16) ? 8 : 0;
#pragma unroll
  for (int v = 0; v < 8; ++v) {
    sOut[(size_t)(mb + v) * HID + n]      = (_Float16)fast_silu(c0[v] + bias);
    sOut[(size_t)(16 + mb + v) * HID + n] = (_Float16)fast_silu(c1[v] + bias);
  }
}

__device__ __forceinline__ void store_f32(float* sOut, int lane, int nt, v8f c0,
                                          v8f c1, float bias) {
  int n = nt * 16 + (lane & 15);
  int mb = (lane & 16) ? 8 : 0;
#pragma unroll
  for (int v = 0; v < 8; ++v) {
    sOut[(size_t)(mb + v) * HID + n]      = c0[v] + bias;
    sOut[(size_t)(16 + mb + v) * HID + n] = c1[v] + bias;
  }
}

enum { M_NODE_ENC = 0, M_EDGE_ENC = 1, M_EDGE_MLP = 2, M_NODE_MLP = 3, M_DECODER = 4 };

struct MlpArgs {
  const float* in_a;     // x / edge_attr / h
  const float* in_b;     // agg (node_mlp)
  const int*   ei;       // edge_index [2][E] (edge_mlp)
  const float* e_feat;   // e (edge_mlp input cols 256..383)
  const _Float16* w0; const float* b0;
  const _Float16* w1; const float* b1;
  const _Float16* w2; const float* b2;
  const float* ln_g; const float* ln_b;
  const float* resid;    // residual (e_old or h), nullable
  float* out;            // [rows,128] f32 (may alias e/h in-place)
  float* agg;            // scatter target (edge_mlp)
  int rows;
};

template <int MODE, int K0T, bool HAS_LN>
__global__ __launch_bounds__(256, 1) void fused_mlp(MlpArgs A) {
  __shared__ _Float16 sA[BROWS * K0T * 32];
  __shared__ _Float16 sH1[BROWS * HID];
  __shared__ _Float16 sH2[BROWS * HID];
  __shared__ float    sLn[BROWS * HID];
  __shared__ float    sMu[BROWS], sRstd[BROWS];
  __shared__ int      sIdx[2][BROWS];  // [0]=src, [1]=tgt

  const int tid  = threadIdx.x;
  const int lane = tid & 31;
  const int nt   = tid >> 5;           // wave id == N-tile
  const int r0   = blockIdx.x * BROWS;
  const int rows = A.rows;

  // ---- stage A into LDS (f16) ----
  if constexpr (MODE == M_EDGE_MLP) {
    if (tid < BROWS) {
      int gr = r0 + tid; if (gr >= rows) gr = rows - 1;
      sIdx[0][tid] = A.ei[gr];                 // src
      sIdx[1][tid] = A.ei[(size_t)rows + gr];  // tgt
    }
    __syncthreads();
    const int K0 = K0T * 32;  // 384
    for (int i = tid; i < BROWS * K0; i += 256) {
      int row = i / K0, col = i - row * K0;
      int gr = r0 + row; if (gr >= rows) gr = rows - 1;
      float v;
      if (col < 128)      v = A.in_a[(size_t)sIdx[1][row] * HID + col];        // x_i = h[tgt]
      else if (col < 256) v = A.in_a[(size_t)sIdx[0][row] * HID + (col - 128)];// x_j = h[src]
      else                v = A.e_feat[(size_t)gr * HID + (col - 256)];
      sA[i] = (_Float16)v;
    }
  } else if constexpr (MODE == M_EDGE_ENC) {
    for (int i = tid; i < BROWS * 32; i += 256) {
      int row = i >> 5, col = i & 31;
      int gr = r0 + row; if (gr >= rows) gr = rows - 1;
      sA[i] = (_Float16)((col < 3) ? A.in_a[(size_t)gr * 3 + col] : 0.f);
    }
  } else if constexpr (MODE == M_NODE_MLP) {
    for (int i = tid; i < BROWS * 256; i += 256) {
      int row = i >> 8, col = i & 255;
      int gr = r0 + row; if (gr >= rows) gr = rows - 1;
      float v = (col < 128) ? A.in_a[(size_t)gr * HID + col]
                            : A.in_b[(size_t)gr * HID + (col - 128)];
      sA[i] = (_Float16)v;
    }
  } else {  // NODE_ENC / DECODER
    for (int i = tid; i < BROWS * HID; i += 256) {
      int row = i >> 7, col = i & 127;
      int gr = r0 + row; if (gr >= rows) gr = rows - 1;
      sA[i] = (_Float16)A.in_a[(size_t)gr * HID + col];
    }
  }
  __syncthreads();

  // ---- layer 0: silu(A @ W0 + b0) ----
  {
    v8f c0 = {}, c1 = {};
    gemm2<K0T>(sA, K0T * 32, A.w0, lane, nt, c0, c1);
    store_silu(sH1, lane, nt, c0, c1, A.b0[nt * 16 + (lane & 15)]);
  }
  __syncthreads();
  // ---- layer 1 ----
  {
    v8f c0 = {}, c1 = {};
    gemm2<4>(sH1, HID, A.w1, lane, nt, c0, c1);
    store_silu(sH2, lane, nt, c0, c1, A.b1[nt * 16 + (lane & 15)]);
  }
  __syncthreads();
  // ---- layer 2 (linear) ----
  {
    v8f c0 = {}, c1 = {};
    gemm2<4>(sH2, HID, A.w2, lane, nt, c0, c1);
    store_f32(sLn, lane, nt, c0, c1, A.b2[nt * 16 + (lane & 15)]);
  }
  __syncthreads();

  // ---- LayerNorm stats (biased var, eps=1e-5) ----
  // 8 threads per row: each reduces 16 elements, then 3-step shfl_xor reduce.
  if constexpr (HAS_LN) {
    {
      int row = tid >> 3, seg = tid & 7;
      const float* p = sLn + (size_t)row * HID + seg * 16;
      float s = 0.f, s2 = 0.f;
#pragma unroll
      for (int c = 0; c < 16; ++c) { float v = p[c]; s += v; s2 += v * v; }
#pragma unroll
      for (int m = 1; m < 8; m <<= 1) {
        s  += __shfl_xor(s, m, 32);
        s2 += __shfl_xor(s2, m, 32);
      }
      if (seg == 0) {
        float mu  = s * (1.f / HID);
        float var = s2 * (1.f / HID) - mu * mu;
        sMu[row]   = mu;
        sRstd[row] = __builtin_amdgcn_rsqf(var + 1e-5f);
      }
    }
    __syncthreads();
  }

  // ---- epilogue: LN, residual, global write, scatter ----
  for (int i = tid; i < BROWS * HID; i += 256) {
    int row = i >> 7, col = i & 127;
    int gr = r0 + row;
    if (gr < rows) {
      float v = sLn[i];
      if constexpr (HAS_LN)
        v = (v - sMu[row]) * sRstd[row] * A.ln_g[col] + A.ln_b[col];
      if constexpr (MODE == M_EDGE_MLP || MODE == M_NODE_MLP)
        v += A.resid[(size_t)gr * HID + col];
      A.out[(size_t)gr * HID + col] = v;
      if constexpr (MODE == M_EDGE_MLP)
        unsafeAtomicAdd(A.agg + (size_t)sIdx[0][row] * HID + col, v);
    }
  }
}

// fp32 [K,128] weight -> f16 WMMA-B-fragment order, K zero-padded to Kp.
// out[((nt*nkt + kt)*32 + lane)*16 + j] = W[k,n], k = kt*32 + (lane&16?16:0) + j,
// n = nt*16 + (lane&15).
__global__ void swizzle_weight_kernel(const float* __restrict__ W,
                                      _Float16* __restrict__ out, int K, int Kp) {
  int idx = blockIdx.x * 256 + threadIdx.x;
  int total = Kp * HID;
  if (idx >= total) return;
  int j = idx & 15;
  int lane = (idx >> 4) & 31;
  int frag = idx >> 9;
  int nkt = Kp >> 5;
  int kt = frag % nkt;
  int nt = frag / nkt;
  int k = kt * 32 + ((lane & 16) ? 16 : 0) + j;
  int n = nt * 16 + (lane & 15);
  out[idx] = (_Float16)((k < K) ? W[(size_t)k * HID + n] : 0.f);
}

__global__ void zero_kernel(float* p, int n) {
  int i = blockIdx.x * 256 + threadIdx.x;
  if (i < n) p[i] = 0.f;
}

// ---------------------------------------------------------------------------
extern "C" void kernel_launch(void* const* d_in, const int* in_sizes, int n_in,
                              void* d_out, int out_size, void* d_ws, size_t ws_size,
                              hipStream_t stream) {
  (void)n_in; (void)out_size; (void)ws_size;
  const int N = in_sizes[0] / HID;   // 50000
  const int E = in_sizes[2] / 3;     // 800000

  const float* x  = (const float*)d_in[0];
  const int*   ei = (const int*)d_in[1];     // [2][E] int32 (JAX x64 off)
  const float* ea = (const float*)d_in[2];

  // params flattened in insertion order: each MLP is w,b,w,b,w,b[,g,b]
  struct MlpP { const float *w0,*b0,*w1,*b1,*w2,*b2,*g,*bt; };
  auto getMlp = [&](int base, bool ln) {
    MlpP p;
    p.w0 = (const float*)d_in[base + 0]; p.b0 = (const float*)d_in[base + 1];
    p.w1 = (const float*)d_in[base + 2]; p.b1 = (const float*)d_in[base + 3];
    p.w2 = (const float*)d_in[base + 4]; p.b2 = (const float*)d_in[base + 5];
    p.g  = ln ? (const float*)d_in[base + 6] : nullptr;
    p.bt = ln ? (const float*)d_in[base + 7] : nullptr;
    return p;
  };
  MlpP node_enc = getMlp(3, true);
  MlpP edge_enc = getMlp(11, true);
  MlpP proc_e[3], proc_n[3];
  for (int l = 0; l < 3; ++l) {
    proc_e[l] = getMlp(19 + 16 * l, true);
    proc_n[l] = getMlp(27 + 16 * l, true);
  }
  MlpP dec = getMlp(19 + 16 * 3, false);  // index 67

  // workspace layout
  size_t off = 0;
  auto alloc = [&](size_t bytes) {
    off = (off + 255) & ~(size_t)255;
    size_t o = off; off += bytes; return o;
  };
  float* h   = (float*)((char*)d_ws + alloc((size_t)N * HID * 4));
  float* agg = (float*)((char*)d_ws + alloc((size_t)N * HID * 4));
  float* e   = (float*)((char*)d_ws + alloc((size_t)E * HID * 4));

  auto swz = [&](const float* w, int K, int Kp) -> const _Float16* {
    _Float16* dst = (_Float16*)((char*)d_ws + alloc((size_t)Kp * HID * 2));
    int total = Kp * HID;
    swizzle_weight_kernel<<<(total + 255) / 256, 256, 0, stream>>>(w, dst, K, Kp);
    return dst;
  };

  struct MlpW { const _Float16 *w0, *w1, *w2; };
  auto swzMlp = [&](const MlpP& p, int K0, int K0p) {
    MlpW w;
    w.w0 = swz(p.w0, K0, K0p);
    w.w1 = swz(p.w1, HID, HID);
    w.w2 = swz(p.w2, HID, HID);
    return w;
  };
  MlpW w_ne = swzMlp(node_enc, HID, HID);
  MlpW w_ee = swzMlp(edge_enc, 3, 32);
  MlpW w_pe[3], w_pn[3];
  for (int l = 0; l < 3; ++l) {
    w_pe[l] = swzMlp(proc_e[l], 384, 384);
    w_pn[l] = swzMlp(proc_n[l], 256, 256);
  }
  MlpW w_de = swzMlp(dec, HID, HID);

  const int nBlk = (N + BROWS - 1) / BROWS;
  const int eBlk = (E + BROWS - 1) / BROWS;

  auto mkArgs = [&](const MlpP& p, const MlpW& w) {
    MlpArgs a{};
    a.w0 = w.w0; a.b0 = p.b0; a.w1 = w.w1; a.b1 = p.b1; a.w2 = w.w2; a.b2 = p.b2;
    a.ln_g = p.g; a.ln_b = p.bt;
    return a;
  };

  // edge encoder: e = MLP_ln(edge_attr)
  {
    MlpArgs a = mkArgs(edge_enc, w_ee);
    a.in_a = ea; a.out = e; a.rows = E;
    fused_mlp<M_EDGE_ENC, 1, true><<<eBlk, 256, 0, stream>>>(a);
  }
  // node encoder: h = MLP_ln(x)
  {
    MlpArgs a = mkArgs(node_enc, w_ne);
    a.in_a = x; a.out = h; a.rows = N;
    fused_mlp<M_NODE_ENC, 4, true><<<nBlk, 256, 0, stream>>>(a);
  }

  for (int l = 0; l < 3; ++l) {
    zero_kernel<<<((N * HID) + 255) / 256, 256, 0, stream>>>(agg, N * HID);
    // e = MLP_ln([h[tgt], h[src], e]) + e ; agg[src] += e_new
    {
      MlpArgs a = mkArgs(proc_e[l], w_pe[l]);
      a.in_a = h; a.ei = ei; a.e_feat = e; a.resid = e; a.out = e; a.agg = agg;
      a.rows = E;
      fused_mlp<M_EDGE_MLP, 12, true><<<eBlk, 256, 0, stream>>>(a);
    }
    // h = h + MLP_ln([h, agg])
    {
      MlpArgs a = mkArgs(proc_n[l], w_pn[l]);
      a.in_a = h; a.in_b = agg; a.resid = h; a.out = h; a.rows = N;
      fused_mlp<M_NODE_MLP, 8, true><<<nBlk, 256, 0, stream>>>(a);
    }
  }

  // decoder (no LN, no residual) -> d_out
  {
    MlpArgs a = mkArgs(dec, w_de);
    a.in_a = h; a.out = (float*)d_out; a.rows = N;
    fused_mlp<M_DECODER, 4, false><<<nBlk, 256, 0, stream>>>(a);
  }
}
